// ImprovedActorCriticNetwork_10385230922201
// MI455X (gfx1250) — matrix operand order repaired and checked by reading, not compiled, not run on hardware
//
#include <hip/hip_runtime.h>
#include <hip/hip_bf16.h>

// ---------------------------------------------------------------------------
// Problem constants (match the reference)
// ---------------------------------------------------------------------------
#define NN    8192      // nodes = U*M
#define EE    131072    // edges
#define HH    8         // heads
#define CC    64        // channels per head
#define HID   512       // H*C
#define UU    16
#define MM    512
#define EMB   64

typedef __attribute__((ext_vector_type(16))) _Float16 v16h;
typedef __attribute__((ext_vector_type(8)))  float    v8f;

union Frag16 { v16h v; uint4 u[2]; };

// ---------------------------------------------------------------------------
// WMMA GEMM:  C[M,Nc] = A[M,K](f16) * Bt[Nc,K](f16, pre-transposed) + bias
// One wave per block; each wave owns a 16(M) x 64(N) tile = 4 accumulators.
// A/B fragments use contiguous k = 16*half + s in both operands (wave32).
// ---------------------------------------------------------------------------
__global__ __launch_bounds__(32)
void wmma_gemm_bias(const _Float16* __restrict__ A,
                    const _Float16* __restrict__ Bt,
                    const float*    __restrict__ bias,
                    float*          __restrict__ C,
                    int K, int Nc) {
  const int lane = threadIdx.x & 31;
  const int l15  = lane & 15;
  const int half = lane >> 4;
  const int m16  = blockIdx.x * 16;
  const int n64  = blockIdx.y * 64;

  v8f acc[4] = {};

  // A: lane row = m16+l15; 16 contiguous K-halves at offset half*16
  const _Float16* arow = A + (size_t)(m16 + l15) * K + half * 16;
  // B: lane column = n64 + j*16 + l15; 16 contiguous K-halves at offset half*16
  const _Float16* brow0 = Bt + (size_t)(n64 +  0 + l15) * K + half * 16;
  const _Float16* brow1 = Bt + (size_t)(n64 + 16 + l15) * K + half * 16;
  const _Float16* brow2 = Bt + (size_t)(n64 + 32 + l15) * K + half * 16;
  const _Float16* brow3 = Bt + (size_t)(n64 + 48 + l15) * K + half * 16;

  for (int k0 = 0; k0 < K; k0 += 32) {
    Frag16 a;
    a.u[0] = *(const uint4*)(arow + k0);
    a.u[1] = *(const uint4*)(arow + k0 + 8);

    Frag16 b;
    b.u[0] = *(const uint4*)(brow0 + k0);
    b.u[1] = *(const uint4*)(brow0 + k0 + 8);
    acc[0] = __builtin_amdgcn_wmma_f32_16x16x32_f16(false, a.v, false, b.v,
                                                    (short)0, acc[0], false, false);
    b.u[0] = *(const uint4*)(brow1 + k0);
    b.u[1] = *(const uint4*)(brow1 + k0 + 8);
    acc[1] = __builtin_amdgcn_wmma_f32_16x16x32_f16(false, a.v, false, b.v,
                                                    (short)0, acc[1], false, false);
    b.u[0] = *(const uint4*)(brow2 + k0);
    b.u[1] = *(const uint4*)(brow2 + k0 + 8);
    acc[2] = __builtin_amdgcn_wmma_f32_16x16x32_f16(false, a.v, false, b.v,
                                                    (short)0, acc[2], false, false);
    b.u[0] = *(const uint4*)(brow3 + k0);
    b.u[1] = *(const uint4*)(brow3 + k0 + 8);
    acc[3] = __builtin_amdgcn_wmma_f32_16x16x32_f16(false, a.v, false, b.v,
                                                    (short)0, acc[3], false, false);
  }

  // D layout: lane l -> n = l&15, rows m = half*8 + r (r = VGPR index 0..7)
  for (int j = 0; j < 4; ++j) {
    const int n  = n64 + j * 16 + l15;
    const float bn = bias[n];
    for (int r = 0; r < 8; ++r) {
      C[(size_t)(m16 + half * 8 + r) * Nc + n] = acc[j][r] + bn;
    }
  }
}

// ---------------------------------------------------------------------------
// Weight transpose + f32->f16:  Wt[n*Kpad + k] = (k < Kact) ? W[k*Nc+n] : 0
// ---------------------------------------------------------------------------
__global__ void transpose_w_f16(const float* __restrict__ W,
                                _Float16* __restrict__ Wt,
                                int Kact, int Kpad, int Nc) {
  int tid = blockIdx.x * blockDim.x + threadIdx.x;
  if (tid >= Nc * Kpad) return;
  int n = tid / Kpad, k = tid - n * Kpad;
  Wt[tid] = (k < Kact) ? (_Float16)W[(size_t)k * Nc + n] : (_Float16)0.f;
}

// ---------------------------------------------------------------------------
// Build layer-0 features directly in f16, K padded 6 -> 32
// x0 = [mc.x, mc.y, mask, speed, dist, ttg, 0...]
// ---------------------------------------------------------------------------
__global__ void build_x0(const float* __restrict__ mc,
                         const float* __restrict__ spd,
                         const float* __restrict__ dist,
                         const float* __restrict__ ttg,
                         const int*   __restrict__ mask,
                         _Float16* __restrict__ xh0) {
  int n = blockIdx.x * blockDim.x + threadIdx.x;
  if (n >= NN) return;
  int u = n >> 9, m = n & 511;
  _Float16* row = xh0 + (size_t)n * 32;
  row[0] = (_Float16)mc[m * 2 + 0];
  row[1] = (_Float16)mc[m * 2 + 1];
  row[2] = (_Float16)(float)mask[u * MM + m];
  row[3] = (_Float16)spd[u];
  row[4] = (_Float16)dist[u * MM + m];
  row[5] = (_Float16)ttg[u * MM + m];
  for (int i = 6; i < 32; ++i) row[i] = (_Float16)0.f;
}

// ---------------------------------------------------------------------------
// Segment-softmax helpers: monotone uint encoding for float atomicMax
// ---------------------------------------------------------------------------
__device__ __forceinline__ unsigned fenc(float x) {
  unsigned u = __float_as_uint(x);
  return (u & 0x80000000u) ? ~u : (u | 0x80000000u);
}
__device__ __forceinline__ float fdec(unsigned k) {
  return __uint_as_float((k & 0x80000000u) ? (k ^ 0x80000000u) : ~k);
}

__global__ void init_seg(unsigned* __restrict__ mxkey, float* __restrict__ denom, int n) {
  int i = blockIdx.x * blockDim.x + threadIdx.x;
  if (i < n) { mxkey[i] = 0u; denom[i] = 0.f; }
}

// score[e,h] = dot(q[dst,h,:], k[src,h,:]) / 8 ; running segment max over dst
__global__ void edge_scores(const float* __restrict__ q,
                            const float* __restrict__ kf,
                            const int*   __restrict__ ei,
                            float*       __restrict__ scores,
                            unsigned*    __restrict__ mxkey) {
  int tid = blockIdx.x * blockDim.x + threadIdx.x;
  if (tid >= EE * HH) return;
  int e = tid >> 3, h = tid & 7;
  int s = ei[e], d = ei[EE + e];
  const float4* qp = (const float4*)(q  + (size_t)d * HID + h * CC);
  const float4* kp = (const float4*)(kf + (size_t)s * HID + h * CC);
  float acc = 0.f;
  for (int i = 0; i < 16; ++i) {
    float4 a = qp[i], b = kp[i];
    acc += a.x * b.x + a.y * b.y + a.z * b.z + a.w * b.w;
  }
  acc *= 0.125f;  // 1/sqrt(64)
  scores[tid] = acc;
  atomicMax(mxkey + (size_t)d * HH + h, fenc(acc));
}

// ex = exp(score - mx[dst]) (in place); denom[dst] += ex
__global__ void edge_exp(float* __restrict__ scores,
                         const int* __restrict__ ei,
                         const unsigned* __restrict__ mxkey,
                         float* __restrict__ denom) {
  int tid = blockIdx.x * blockDim.x + threadIdx.x;
  if (tid >= EE * HH) return;
  int e = tid >> 3, h = tid & 7;
  int d = ei[EE + e];
  float ex = expf(scores[tid] - fdec(mxkey[(size_t)d * HH + h]));
  scores[tid] = ex;
  atomicAdd(denom + (size_t)d * HH + h, ex);
}

// agg[dst, c..c+3] += alpha[e,h] * v[src, c..c+3]   (agg pre-loaded with skip)
__global__ void edge_scatter(const float* __restrict__ ex,
                             const float* __restrict__ denom,
                             const float* __restrict__ v,
                             const int*   __restrict__ ei,
                             float*       __restrict__ agg) {
  int tid = blockIdx.x * blockDim.x + threadIdx.x;
  if (tid >= EE * (HID / 4)) return;
  int e  = tid >> 7;
  int c4 = tid & 127;
  int c  = c4 << 2;
  int h  = c >> 6;
  int s = ei[e], d = ei[EE + e];
  float alpha = ex[(size_t)e * HH + h] / (denom[(size_t)d * HH + h] + 1e-16f);
  float4 vv = *(const float4*)(v + (size_t)s * HID + c);
  float* o = agg + (size_t)d * HID + c;
  atomicAdd(o + 0, alpha * vv.x);
  atomicAdd(o + 1, alpha * vv.y);
  atomicAdd(o + 2, alpha * vv.z);
  atomicAdd(o + 3, alpha * vv.w);
}

// next-layer features: xh = f16(relu(agg))
__global__ void relu_to_half(const float* __restrict__ agg,
                             _Float16* __restrict__ xh, int n) {
  int i = blockIdx.x * blockDim.x + threadIdx.x;
  if (i < n) xh[i] = (_Float16)fmaxf(agg[i], 0.f);
}

// uav_emb[u,c] = sum_m xout[u*M+m, c]
__global__ void pool_sum(const float* __restrict__ xout, float* __restrict__ emb) {
  int u = blockIdx.x, c = threadIdx.x;   // 16 blocks x 64 threads
  const float* p = xout + (size_t)u * MM * EMB + c;
  float s = 0.f;
  for (int m = 0; m < MM; ++m) s += p[(size_t)m * EMB];
  emb[u * EMB + c] = s;
}

// Actor/critic heads. combined = [uav_info(2), emb(64), speed(1)] (67 dims).
// softmax over a [U,1] logit axis is identically 1.0.
__global__ __launch_bounds__(128)
void heads(const float* __restrict__ uinfo, const float* __restrict__ spd,
           const float* __restrict__ emb,
           const float* __restrict__ Wa1, const float* __restrict__ ba1,
           const float* __restrict__ Wa2, const float* __restrict__ ba2,
           const float* __restrict__ Wc1, const float* __restrict__ bc1,
           const float* __restrict__ Wc2, const float* __restrict__ bc2,
           float* __restrict__ out) {
  __shared__ float sm[128];
  int t = threadIdx.x;
  for (int u = 0; u < UU; ++u) {
    float ca = uinfo[u * 2 + 0] * Wa1[0 * 128 + t] + uinfo[u * 2 + 1] * Wa1[1 * 128 + t];
    float cc = uinfo[u * 2 + 0] * Wc1[0 * 128 + t] + uinfo[u * 2 + 1] * Wc1[1 * 128 + t];
    for (int c = 0; c < EMB; ++c) {
      float e = emb[u * EMB + c];
      ca += e * Wa1[(2 + c) * 128 + t];
      cc += e * Wc1[(2 + c) * 128 + t];
    }
    ca += spd[u] * Wa1[66 * 128 + t] + ba1[t];
    cc += spd[u] * Wc1[66 * 128 + t] + bc1[t];
    ca = fmaxf(ca, 0.f) * Wa2[t];
    cc = fmaxf(cc, 0.f) * Wc2[t];

    sm[t] = ca; __syncthreads();
    for (int o = 64; o > 0; o >>= 1) { if (t < o) sm[t] += sm[t + o]; __syncthreads(); }
    float logit_a = sm[0] + ba2[0];   // softmax([x]) == [1.0]
    (void)logit_a;
    __syncthreads();

    sm[t] = cc; __syncthreads();
    for (int o = 64; o > 0; o >>= 1) { if (t < o) sm[t] += sm[t + o]; __syncthreads(); }
    if (t == 0) {
      out[u]      = 1.0f;              // action_probs
      out[UU + u] = sm[0] + bc2[0];    // state_values
    }
    __syncthreads();
  }
}

// ---------------------------------------------------------------------------
// Host launcher
// ---------------------------------------------------------------------------
extern "C" void kernel_launch(void* const* d_in, const int* in_sizes, int n_in,
                              void* d_out, int out_size, void* d_ws, size_t ws_size,
                              hipStream_t stream) {
  (void)in_sizes; (void)n_in; (void)out_size; (void)ws_size;

  const float* mc    = (const float*)d_in[0];
  const float* uinfo = (const float*)d_in[1];
  const float* spd   = (const float*)d_in[2];
  const float* dist  = (const float*)d_in[3];
  const float* ttg   = (const float*)d_in[4];
  const int*   mask  = (const int*)  d_in[5];
  const int*   ei    = (const int*)  d_in[6];   // [2, E] int32
  const float* Wq0 = (const float*)d_in[8],  *bq0 = (const float*)d_in[9];
  const float* Wk0 = (const float*)d_in[10], *bk0 = (const float*)d_in[11];
  const float* Wv0 = (const float*)d_in[12], *bv0 = (const float*)d_in[13];
  const float* Ws0 = (const float*)d_in[14], *bs0 = (const float*)d_in[15];
  const float* Wq  = (const float*)d_in[16], *bq  = (const float*)d_in[17];
  const float* Wk  = (const float*)d_in[18], *bk  = (const float*)d_in[19];
  const float* Wv  = (const float*)d_in[20], *bv  = (const float*)d_in[21];
  const float* Ws  = (const float*)d_in[22], *bs  = (const float*)d_in[23];
  const float* Wout = (const float*)d_in[24], *bout = (const float*)d_in[25];
  const float* Wa1 = (const float*)d_in[26], *ba1 = (const float*)d_in[27];
  const float* Wa2 = (const float*)d_in[28], *ba2 = (const float*)d_in[29];
  const float* Wc1 = (const float*)d_in[30], *bc1 = (const float*)d_in[31];
  const float* Wc2 = (const float*)d_in[32], *bc2 = (const float*)d_in[33];

  // workspace carve-up
  char* ws = (char*)d_ws;
  size_t off = 0;
  auto alloc = [&](size_t bytes) -> void* {
    void* p = ws + off;
    off = (off + bytes + 255) & ~(size_t)255;
    return p;
  };
  _Float16* xh0    = (_Float16*)alloc((size_t)NN * 32 * 2);
  _Float16* xh     = (_Float16*)alloc((size_t)NN * HID * 2);
  _Float16* Wt     = (_Float16*)alloc((size_t)HID * HID * 2);
  float*    qf     = (float*)   alloc((size_t)NN * HID * 4);
  float*    kf     = (float*)   alloc((size_t)NN * HID * 4);
  float*    vf     = (float*)   alloc((size_t)NN * HID * 4);
  float*    agg    = (float*)   alloc((size_t)NN * HID * 4);
  float*    scores = (float*)   alloc((size_t)EE * HH * 4);
  unsigned* mxkey  = (unsigned*)alloc((size_t)NN * HH * 4);
  float*    denom  = (float*)   alloc((size_t)NN * HH * 4);
  float*    xout   = (float*)   alloc((size_t)NN * EMB * 4);
  float*    emb    = (float*)   alloc((size_t)UU * EMB * 4);

  build_x0<<<(NN + 255) / 256, 256, 0, stream>>>(mc, spd, dist, ttg, mask, xh0);

  const dim3 gGemm(NN / 16, HID / 64);

  for (int l = 0; l < 4; ++l) {
    const float *Wq_, *bq_, *Wk_, *bk_, *Wv_, *bv_, *Ws_, *bs_;
    int Kact, Kpad;
    const _Float16* Axh;
    if (l == 0) {
      Wq_ = Wq0; bq_ = bq0; Wk_ = Wk0; bk_ = bk0;
      Wv_ = Wv0; bv_ = bv0; Ws_ = Ws0; bs_ = bs0;
      Kact = 6; Kpad = 32; Axh = xh0;
    } else {
      size_t wo = (size_t)(l - 1) * HID * HID, bo = (size_t)(l - 1) * HID;
      Wq_ = Wq + wo; bq_ = bq + bo; Wk_ = Wk + wo; bk_ = bk + bo;
      Wv_ = Wv + wo; bv_ = bv + bo; Ws_ = Ws + wo; bs_ = bs + bo;
      Kact = HID; Kpad = HID; Axh = xh;
    }
    const int tThreads = HID * Kpad;
    const int tBlocks  = (tThreads + 255) / 256;

    transpose_w_f16<<<tBlocks, 256, 0, stream>>>(Wq_, Wt, Kact, Kpad, HID);
    wmma_gemm_bias<<<gGemm, 32, 0, stream>>>(Axh, Wt, bq_, qf, Kpad, HID);
    transpose_w_f16<<<tBlocks, 256, 0, stream>>>(Wk_, Wt, Kact, Kpad, HID);
    wmma_gemm_bias<<<gGemm, 32, 0, stream>>>(Axh, Wt, bk_, kf, Kpad, HID);
    transpose_w_f16<<<tBlocks, 256, 0, stream>>>(Wv_, Wt, Kact, Kpad, HID);
    wmma_gemm_bias<<<gGemm, 32, 0, stream>>>(Axh, Wt, bv_, vf, Kpad, HID);
    // skip connection written straight into the aggregation buffer
    transpose_w_f16<<<tBlocks, 256, 0, stream>>>(Ws_, Wt, Kact, Kpad, HID);
    wmma_gemm_bias<<<gGemm, 32, 0, stream>>>(Axh, Wt, bs_, agg, Kpad, HID);

    init_seg<<<(NN * HH + 255) / 256, 256, 0, stream>>>(mxkey, denom, NN * HH);
    edge_scores<<<(EE * HH + 255) / 256, 256, 0, stream>>>(qf, kf, ei, scores, mxkey);
    edge_exp<<<(EE * HH + 255) / 256, 256, 0, stream>>>(scores, ei, mxkey, denom);
    edge_scatter<<<(EE * (HID / 4) + 255) / 256, 256, 0, stream>>>(scores, denom, vf, ei, agg);
    relu_to_half<<<(NN * HID + 255) / 256, 256, 0, stream>>>(agg, xh, NN * HID);
  }

  // output projection [8192,512] x [512,64]
  transpose_w_f16<<<(EMB * HID + 255) / 256, 256, 0, stream>>>(Wout, Wt, HID, HID, EMB);
  wmma_gemm_bias<<<dim3(NN / 16, EMB / 64), 32, 0, stream>>>(xh, Wt, bout, xout, HID, EMB);

  pool_sum<<<UU, EMB, 0, stream>>>(xout, emb);
  heads<<<1, 128, 0, stream>>>(uinfo, spd, emb, Wa1, ba1, Wa2, ba2,
                               Wc1, bc1, Wc2, bc2, (float*)d_out);
}
